// SGN_58763742544553
// MI455X (gfx1250) — matrix operand order repaired and checked by reading, not compiled, not run on hardware
//
#include <hip/hip_runtime.h>
#include <stdint.h>

typedef __attribute__((ext_vector_type(16))) _Float16 v16h;
typedef __attribute__((ext_vector_type(8)))  float    v8f;

#define NSAMP 512
#define SEG   20
#define NJ    25

// ---- input pointer order: setup_inputs() dict order, depth-first ----
enum {
  IN_X = 0,
  J_BN_G, J_BN_B, J_BN_M, J_BN_V, J_W1, J_B1, J_W2, J_B2,
  D_BN_G, D_BN_B, D_BN_M, D_BN_V, D_W1, D_B1, D_W2, D_B2,
  S_W1, S_B1, S_W2, S_B2,
  T_W1, T_B1, T_W2, T_B2,
  G_INW, G_INB, G_INUW, G_INUB, G_ALPHAS, G_ATT0,
  G_OUTW, G_OUTB, G_OUTBN_G, G_OUTBN_B, G_OUTBN_M, G_OUTBN_V,
  G_FFW, G_FFB, G_FFBN_G, G_FFBN_B, G_FFBN_M, G_FFBN_V,
  G_D1W, G_D1B, G_D1BN_G, G_D1BN_B, G_D1BN_M, G_D1BN_V,
  G_D2W, G_D2B, G_D2BN_G, G_D2BN_B, G_D2BN_M, G_D2BN_V,
  G_OTW, G_OTB, G_OTBN_G, G_OTBN_B, G_OTBN_M, G_OTBN_V,
  G_DT2W, G_DT2B, G_DT2BN_G, G_DT2BN_B, G_DT2BN_M, G_DT2BN_V,
  C_C1W, C_C1B, C_BN1_G, C_BN1_B, C_BN1_M, C_BN1_V,
  C_C2W, C_C2B, C_BN2_G, C_BN2_B, C_BN2_M, C_BN2_V,
  FC_W, FC_B,
  N_IN
};

struct P { const float* a[N_IN]; };

// ---------------- WMMA helpers (CDNA5 16x16x32 f16 -> f32) ----------------
__device__ __forceinline__ v8f wmma16(v16h a, v16h b, v8f c) {
  return __builtin_amdgcn_wmma_f32_16x16x32_f16(false, a, false, b, (short)0, c,
                                                false, false);
}

// A matrix 16x32 f16, row-major source with leading dim `ld`.
// lanes 0-15: row m0+lane, K elems e<8 -> k0+e, e>=8 -> k0+e+8
// lanes 16-31: same rows? no: row m0+(lane&15), K base +8
__device__ __forceinline__ v16h ld_a(const _Float16* base, int ld, int m0,
                                     int k0, int lane) {
  int m  = m0 + (lane & 15);
  int kb = k0 + ((lane & 16) ? 8 : 0);
  v16h a;
#pragma unroll
  for (int e = 0; e < 16; ++e) a[e] = base[m * ld + kb + e + ((e < 8) ? 0 : 8)];
  return a;
}

// B matrix 32x16 f16, row-major (K x N) with leading dim `ld`.
__device__ __forceinline__ v16h ld_b(const _Float16* base, int ld, int k0,
                                     int n0, int lane) {
  int ncur = n0 + (lane & 15);
  int kb   = k0 + ((lane & 16) ? 8 : 0);
  v16h b;
#pragma unroll
  for (int e = 0; e < 16; ++e) b[e] = base[(kb + e + ((e < 8) ? 0 : 8)) * ld + ncur];
  return b;
}

// A tile of the unfolded temporal window: A[c][u], u = w*25+v, window w-1 shift.
__device__ __forceinline__ v16h ld_a_xu(const _Float16* xh, int t, int m0,
                                        int k0, int lane) {
  int c  = m0 + (lane & 15);
  int kb = k0 + ((lane & 16) ? 8 : 0);
  v16h a;
#pragma unroll
  for (int e = 0; e < 16; ++e) {
    int u = kb + e + ((e < 8) ? 0 : 8);
    _Float16 val = (_Float16)0.f;
    if (u < 75) {
      int w  = u / 25;
      int vv = u - w * 25;
      int tt = t + w - 1;
      if (tt >= 0 && tt < SEG) val = xh[c * 640 + tt * 32 + vv];
    }
    a[e] = val;
  }
  return a;
}

__device__ __forceinline__ void st_c_f32(float* base, int ld, int m0, int n0,
                                         int lane, v8f c) {
  int ncur = n0 + (lane & 15);
  int mb   = m0 + ((lane & 16) ? 8 : 0);
#pragma unroll
  for (int e = 0; e < 8; ++e) base[(mb + e) * ld + ncur] = c[e];
}

__device__ __forceinline__ void st_c_f16(_Float16* base, int ld, int m0, int n0,
                                         int lane, v8f c) {
  int ncur = n0 + (lane & 15);
  int mb   = m0 + ((lane & 16) ? 8 : 0);
#pragma unroll
  for (int e = 0; e < 8; ++e) base[(mb + e) * ld + ncur] = (_Float16)c[e];
}

__device__ __forceinline__ float lrelu(float x) { return x >= 0.f ? x : 0.1f * x; }

// ---------------- LDS layout for the GAT kernel (bytes) ----------------
#define XH_OFF   0         // f16 [32][640]  (c, t*32+v)           40960
#define ATTN_OFF 40960     // f16 [8][96][32]                      49152
#define U_OFF    90112     // phase-unioned region                 69632
#define WSC_OFF  159744    // f16 [8 waves][64][32]                32768
#define YBUF_OFF 192512    // f16 [64][640]                        81920
#define MJ_OFF   274432    // f32 [64][20]                          5120
#define PRM_OFF  279552    // f32 bn scale/shift (6x128) + alphas   4096
#define T1_OFF   283648    // f32 tem1 [64][20]                     5120
#define C1O_OFF  288768    // f32 [64][20]                          5120
#define C2O_OFF  293888    // f32 [128][20]                        10240
#define M2_OFF   304128    // f32 [128]                              512
#define SMEM_TOTAL 304640
// union U: phase A: qs f16[80][160]@+0, ks f16[160][32]@+25600,
//          sco f32[80][32]@+35840, winF f32[2048]@+46080, winuF@+54272
//      phase B: Wout f16[64][256]@+0, Wd1@+32768, Wff@+36864, Wd2@+45056
//      phase C: Wot f16[64][448]@+0, Wdt2 f16[64][64]@+57344

#define PK_OUT 0
#define PK_D1  1
#define PK_FF  2
#define PK_D2  3
#define PK_OT  4
#define PK_DT2 5

// =================== kernel 1: spa1 / tem1 (one-hot embeds) ===================
__global__ void k_precompute(P p, float* ws) {
  int tid = threadIdx.x;
  if (tid < NJ) {  // spa1[16][25] at ws[0..400)
    int j = tid;
    float acc[16];
#pragma unroll
    for (int c = 0; c < 16; ++c) acc[c] = 0.f;
    for (int o = 0; o < 64; ++o) {
      float h = fmaxf(p.a[S_W1][o * NJ + j] + p.a[S_B1][o], 0.f);
      for (int c = 0; c < 16; ++c) acc[c] += p.a[S_W2][c * 64 + o] * h;
    }
    for (int c = 0; c < 16; ++c)
      ws[c * NJ + j] = fmaxf(acc[c] + p.a[S_B2][c], 0.f);
  }
  if (tid >= 32 && tid < 32 + SEG) {  // tem1[64][20] at ws[512..1792)
    int t = tid - 32;
    float acc[64];
    for (int c = 0; c < 64; ++c) acc[c] = 0.f;
    for (int o = 0; o < 64; ++o) {
      float h = fmaxf(p.a[T_W1][o * SEG + t] + p.a[T_B1][o], 0.f);
      for (int c = 0; c < 64; ++c) acc[c] += p.a[T_W2][c * 64 + o] * h;
    }
    for (int c = 0; c < 64; ++c)
      ws[512 + c * SEG + t] = fmaxf(acc[c] + p.a[T_B2][c], 0.f);
  }
}

// =================== kernel 2: joint/dif embed -> h0 f16 ===================
__global__ void __launch_bounds__(512) k_embed(P p, const float* ws,
                                               _Float16* h0) {
  __shared__ float jw1[192], jb1[64], jw2[1024], jb2[16], jsc[75], jsh[75];
  __shared__ float dw1[192], db1[64], dw2[1024], db2[16], dsc[75], dsh[75];
  __shared__ float spaL[400];
  int n = blockIdx.x, tid = threadIdx.x;
  for (int i = tid; i < 192; i += 512) { jw1[i] = p.a[J_W1][i]; dw1[i] = p.a[D_W1][i]; }
  for (int i = tid; i < 64; i += 512) { jb1[i] = p.a[J_B1][i]; db1[i] = p.a[D_B1][i]; }
  for (int i = tid; i < 1024; i += 512) { jw2[i] = p.a[J_W2][i]; dw2[i] = p.a[D_W2][i]; }
  for (int i = tid; i < 16; i += 512) { jb2[i] = p.a[J_B2][i]; db2[i] = p.a[D_B2][i]; }
  for (int i = tid; i < 75; i += 512) {
    float sc = p.a[J_BN_G][i] * rsqrtf(p.a[J_BN_V][i] + 1e-5f);
    jsc[i] = sc; jsh[i] = p.a[J_BN_B][i] - p.a[J_BN_M][i] * sc;
    float sd = p.a[D_BN_G][i] * rsqrtf(p.a[D_BN_V][i] + 1e-5f);
    dsc[i] = sd; dsh[i] = p.a[D_BN_B][i] - p.a[D_BN_M][i] * sd;
  }
  for (int i = tid; i < 400; i += 512) spaL[i] = ws[i];
  // zero this sample's slab (incl. v-padding)
  uint32_t* slab32 = (uint32_t*)(h0 + (size_t)n * 32 * 640);
  for (int i = tid; i < 32 * 640 / 2; i += 512) slab32[i] = 0u;
  __syncthreads();

  _Float16* slab = h0 + (size_t)n * 32 * 640;
  for (int idx = tid; idx < NJ * SEG; idx += 512) {
    int j = idx / SEG, t = idx % SEG;
    const float* xr = p.a[IN_X] + (size_t)n * 1500 + t * 75 + j * 3;
    float pb[3], db[3];
#pragma unroll
    for (int c = 0; c < 3; ++c) {
      float pc = xr[c];
      pb[c] = pc * jsc[c * NJ + j] + jsh[c * NJ + j];
      float dv = (t > 0) ? (pc - xr[c - 75]) : 0.f;
      db[c] = dv * dsc[c * NJ + j] + dsh[c * NJ + j];
    }
    float a1[16], a2[16];
#pragma unroll
    for (int c = 0; c < 16; ++c) { a1[c] = 0.f; a2[c] = 0.f; }
    for (int o = 0; o < 64; ++o) {
      float h = fmaxf(jw1[o * 3] * pb[0] + jw1[o * 3 + 1] * pb[1] +
                      jw1[o * 3 + 2] * pb[2] + jb1[o], 0.f);
      float hd = fmaxf(dw1[o * 3] * db[0] + dw1[o * 3 + 1] * db[1] +
                       dw1[o * 3 + 2] * db[2] + db1[o], 0.f);
      for (int c = 0; c < 16; ++c) {
        a1[c] += jw2[c * 64 + o] * h;
        a2[c] += dw2[c * 64 + o] * hd;
      }
    }
#pragma unroll
    for (int c = 0; c < 16; ++c) {
      float v = fmaxf(a1[c] + jb2[c], 0.f) + fmaxf(a2[c] + db2[c], 0.f);
      slab[c * 640 + t * 32 + j] = (_Float16)v;
      slab[(16 + c) * 640 + t * 32 + j] = (_Float16)spaL[c * NJ + j];
    }
  }
}

// =================== kernel 3: full GAT + tail, one block per sample ========
__global__ void __launch_bounds__(256) k_gat(P p, const float* ws,
                                             const _Float16* h0, float* out) {
  extern __shared__ char smem[];
  _Float16* xh    = (_Float16*)(smem + XH_OFF);
  _Float16* attnL = (_Float16*)(smem + ATTN_OFF);
  _Float16* wsc   = (_Float16*)(smem + WSC_OFF);
  _Float16* ybuf  = (_Float16*)(smem + YBUF_OFF);
  float*    mj    = (float*)(smem + MJ_OFF);
  float*    prm   = (float*)(smem + PRM_OFF);
  float*    t1    = (float*)(smem + T1_OFF);
  // phase A union views
  _Float16* qs    = (_Float16*)(smem + U_OFF);
  _Float16* ks    = (_Float16*)(smem + U_OFF + 25600);
  float*    sco   = (float*)(smem + U_OFF + 35840);
  float*    winF  = (float*)(smem + U_OFF + 46080);
  float*    winuF = (float*)(smem + U_OFF + 54272);
  // phase B union views
  _Float16* WoutL = (_Float16*)(smem + U_OFF);
  _Float16* Wd1L  = (_Float16*)(smem + U_OFF + 32768);
  _Float16* WffL  = (_Float16*)(smem + U_OFF + 36864);
  _Float16* Wd2L  = (_Float16*)(smem + U_OFF + 45056);
  // phase C union views
  _Float16* WotL  = (_Float16*)(smem + U_OFF);
  _Float16* Wdt2L = (_Float16*)(smem + U_OFF + 57344);

  int n = blockIdx.x, tid = threadIdx.x;
  int lane = tid & 31, wave = tid >> 5;
  v8f zv = {};

  if (tid == 0) {  // CDNA5 global_prefetch_b8 hints for hot weights
    __builtin_prefetch(p.a[G_OUTW], 0, 1);
    __builtin_prefetch(p.a[G_OTW], 0, 1);
    __builtin_prefetch(p.a[G_ATT0], 0, 1);
  }

  // ---- stage x, in/inu weights, bn scale/shift (+bias folded), tem1 ----
  {
    const uint32_t* src = (const uint32_t*)(h0 + (size_t)n * 32 * 640);
    uint32_t* dst = (uint32_t*)xh;
    for (int i = tid; i < 32 * 640 / 2; i += 256) dst[i] = src[i];
  }
  for (int i = tid; i < 2048; i += 256) {
    winF[i] = p.a[G_INW][i];
    winuF[i] = p.a[G_INUW][i];
  }
  if (tid < 64) {
    int o = tid;
    const int gi[6] = {G_OUTBN_G, G_D1BN_G, G_FFBN_G, G_D2BN_G, G_OTBN_G, G_DT2BN_G};
    const int bi[6] = {G_OUTB,    G_D1B,    G_FFB,    G_D2B,    G_OTB,    G_DT2B};
#pragma unroll
    for (int k = 0; k < 6; ++k) {
      float g = p.a[gi[k]][o], b = p.a[gi[k] + 1][o];
      float m = p.a[gi[k] + 2][o], vv = p.a[gi[k] + 3][o];
      float sc = g * rsqrtf(vv + 1e-5f);
      prm[k * 128 + o] = sc;
      prm[k * 128 + 64 + o] = (p.a[bi[k]][o] - m) * sc + b;  // bias folded
    }
  }
  if (tid >= 64 && tid < 72) prm[768 + tid - 64] = p.a[G_ALPHAS][tid - 64];
  for (int i = tid; i < 64 * SEG; i += 256) t1[i] = ws[512 + i];
  __syncthreads();

  // ================= Phase A: attention for all 8 subsets =================
  for (int s = 0; s < 8; ++s) {
    // q_s^T : [u=80][ct=160]   (zero-padded rows u>=75; bias on every row u<75)
    for (int idx = tid; idx < 80 * 160; idx += 256) {
      int u = idx / 160, ct = idx % 160;
      int c = ct / SEG, t = ct % SEG;
      float acc = 0.f;
      if (u < 75) {
        acc = p.a[G_INUB][8 * s + c];  // 1x1 conv bias applies everywhere (pad is pre-conv)
        int w = u / 25, vv = u - w * 25, tt = t + w - 1;
        if (tt >= 0 && tt < SEG) {
          const float* wr = winuF + (8 * s + c) * 32;
          const _Float16* xr = xh + tt * 32 + vv;
          for (int i = 0; i < 32; ++i) acc += wr[i] * (float)xr[i * 640];
        }
      }
      qs[idx] = (_Float16)acc;
    }
    // k_s : [ct=160][v=32]
    for (int idx = tid; idx < 160 * 32; idx += 256) {
      int ct = idx >> 5, vv = idx & 31;
      int c = ct / SEG, t = ct % SEG;
      const float* wr = winF + (8 * s + c) * 32;
      const _Float16* xr = xh + t * 32 + vv;
      float acc = p.a[G_INB][8 * s + c];
      for (int i = 0; i < 32; ++i) acc += wr[i] * (float)xr[i * 640];
      ks[idx] = (_Float16)acc;
    }
    __syncthreads();
    // scores = qs^T . ks / 160  : M=80, N=32, K=160 (WMMA, 10 tiles over 8 waves)
    for (int tile = wave; tile < 10; tile += 8) {
      int mt = tile >> 1, nt = tile & 1;
      v8f acc = zv;
      for (int kk = 0; kk < 5; ++kk) {
        v16h a = ld_a(qs, 160, mt * 16, kk * 32, lane);
        v16h b = ld_b(ks, 32, kk * 32, nt * 16, lane);
        acc = wmma16(a, b, acc);
      }
#pragma unroll
      for (int e = 0; e < 8; ++e) acc[e] *= (1.f / 160.f);
      st_c_f32(sco, 32, mt * 16, nt * 16, lane, acc);
    }
    __syncthreads();
    // softmax over u (75 valid), attn = sm*alpha + att0 ; pad rows/cols = 0
    if (tid < 32) {
      int vv = tid;
      float alpha = prm[768 + s];
      _Float16* ac = attnL + s * 96 * 32 + vv;
      if (vv < NJ) {
        float mx = -3.4e38f;
        for (int u = 0; u < 75; ++u) mx = fmaxf(mx, sco[u * 32 + vv]);
        float ssum = 0.f;
        for (int u = 0; u < 75; ++u) ssum += __expf(sco[u * 32 + vv] - mx);
        float inv = 1.f / ssum;
        for (int u = 0; u < 75; ++u) {
          float a0 = p.a[G_ATT0][(s * 75 + u) * NJ + vv];
          ac[u * 32] = (_Float16)(__expf(sco[u * 32 + vv] - mx) * inv * alpha + a0);
        }
        for (int u = 75; u < 96; ++u) ac[u * 32] = (_Float16)0.f;
      } else {
        for (int u = 0; u < 96; ++u) ac[u * 32] = (_Float16)0.f;
      }
    }
    __syncthreads();
  }

  // ================= Phase B: attn apply + out/d1/ff/d2 =================
  for (int i = tid; i < 16384; i += 256) WoutL[i] = (_Float16)p.a[G_OUTW][i];
  for (int i = tid; i < 2048; i += 256) {
    Wd1L[i] = (_Float16)p.a[G_D1W][i];
    Wd2L[i] = (_Float16)p.a[G_D2W][i];
  }
  for (int i = tid; i < 4096; i += 256) WffL[i] = (_Float16)p.a[G_FFW][i];
  __syncthreads();

  _Float16* myw = wsc + wave * 2048;  // [64][32] per-wave scratch
  int hi8 = (lane & 16) ? 8 : 0;
  int col = lane & 15;

  for (int t = wave; t < SEG; t += 8) {
    v8f acc[4][2];
#pragma unroll
    for (int mt = 0; mt < 4; ++mt) { acc[mt][0] = zv; acc[mt][1] = zv; }
    for (int s = 0; s < 8; ++s) {
      // y_s = Xu_t (32x96) . attn_s (96x32)
      v8f yt[2][2];
      yt[0][0] = zv; yt[0][1] = zv; yt[1][0] = zv; yt[1][1] = zv;
      const _Float16* at = attnL + s * 96 * 32;
      for (int kk = 0; kk < 3; ++kk) {
        v16h b0 = ld_b(at, 32, kk * 32, 0, lane);
        v16h b1 = ld_b(at, 32, kk * 32, 16, lane);
#pragma unroll
        for (int mt = 0; mt < 2; ++mt) {
          v16h a = ld_a_xu(xh, t, mt * 16, kk * 32, lane);
          yt[mt][0] = wmma16(a, b0, yt[mt][0]);
          yt[mt][1] = wmma16(a, b1, yt[mt][1]);
        }
      }
#pragma unroll
      for (int mt = 0; mt < 2; ++mt) {
        st_c_f16(myw, 32, mt * 16, 0, lane, yt[mt][0]);
        st_c_f16(myw, 32, mt * 16, 16, lane, yt[mt][1]);
      }
      // out += Wout[:, 32s:32s+32] . y_s
      {
        v16h b0 = ld_b(myw, 32, 0, 0, lane);
        v16h b1 = ld_b(myw, 32, 0, 16, lane);
#pragma unroll
        for (int mt = 0; mt < 4; ++mt) {
          v16h a = ld_a(WoutL, 256, mt * 16, s * 32, lane);
          acc[mt][0] = wmma16(a, b0, acc[mt][0]);
          acc[mt][1] = wmma16(a, b1, acc[mt][1]);
        }
      }
    }
    // d1 = Wd1 . x_t
    v8f dacc[4][2];
#pragma unroll
    for (int mt = 0; mt < 4; ++mt) { dacc[mt][0] = zv; dacc[mt][1] = zv; }
    {
      v16h b0 = ld_b(xh + t * 32, 640, 0, 0, lane);
      v16h b1 = ld_b(xh + t * 32, 640, 0, 16, lane);
#pragma unroll
      for (int mt = 0; mt < 4; ++mt) {
        v16h a = ld_a(Wd1L, 32, mt * 16, 0, lane);
        dacc[mt][0] = wmma16(a, b0, dacc[mt][0]);
        dacc[mt][1] = wmma16(a, b1, dacc[mt][1]);
      }
    }
    // yA = leaky(bn_d1(d1) + bn_out(out)) -> myw[64][32]
#pragma unroll
    for (int mt = 0; mt < 4; ++mt)
#pragma unroll
      for (int nt = 0; nt < 2; ++nt)
#pragma unroll
        for (int e = 0; e < 8; ++e) {
          int o = mt * 16 + hi8 + e;
          float ov = acc[mt][nt][e] * prm[PK_OUT * 128 + o] + prm[PK_OUT * 128 + 64 + o];
          float dv = dacc[mt][nt][e] * prm[PK_D1 * 128 + o] + prm[PK_D1 * 128 + 64 + o];
          myw[o * 32 + nt * 16 + col] = (_Float16)lrelu(ov + dv);
        }
    // ff = Wff . yA ; d2 = Wd2 . x_t
    v8f facc[4][2];
#pragma unroll
    for (int mt = 0; mt < 4; ++mt) { facc[mt][0] = zv; facc[mt][1] = zv; }
    for (int kk = 0; kk < 2; ++kk) {
      v16h b0 = ld_b(myw, 32, kk * 32, 0, lane);
      v16h b1 = ld_b(myw, 32, kk * 32, 16, lane);
#pragma unroll
      for (int mt = 0; mt < 4; ++mt) {
        v16h a = ld_a(WffL, 64, mt * 16, kk * 32, lane);
        facc[mt][0] = wmma16(a, b0, facc[mt][0]);
        facc[mt][1] = wmma16(a, b1, facc[mt][1]);
      }
    }
#pragma unroll
    for (int mt = 0; mt < 4; ++mt) { dacc[mt][0] = zv; dacc[mt][1] = zv; }
    {
      v16h b0 = ld_b(xh + t * 32, 640, 0, 0, lane);
      v16h b1 = ld_b(xh + t * 32, 640, 0, 16, lane);
#pragma unroll
      for (int mt = 0; mt < 4; ++mt) {
        v16h a = ld_a(Wd2L, 32, mt * 16, 0, lane);
        dacc[mt][0] = wmma16(a, b0, dacc[mt][0]);
        dacc[mt][1] = wmma16(a, b1, dacc[mt][1]);
      }
    }
    // yB = leaky(bn_d2(d2) + bn_ff(ff)) -> ybuf[:, t, :]
#pragma unroll
    for (int mt = 0; mt < 4; ++mt)
#pragma unroll
      for (int nt = 0; nt < 2; ++nt)
#pragma unroll
        for (int e = 0; e < 8; ++e) {
          int o = mt * 16 + hi8 + e;
          float fv = facc[mt][nt][e] * prm[PK_FF * 128 + o] + prm[PK_FF * 128 + 64 + o];
          float dv = dacc[mt][nt][e] * prm[PK_D2 * 128 + o] + prm[PK_D2 * 128 + 64 + o];
          ybuf[o * 640 + t * 32 + nt * 16 + col] = (_Float16)lrelu(dv + fv);
        }
  }
  __syncthreads();

  // ================= Phase C: 7x1 temporal conv + dt2, joint max ==========
  for (int e = tid; e < 64 * 448; e += 256) {
    int o = e / 448, r = e % 448, dt = r >> 6, i = r & 63;
    WotL[e] = (_Float16)p.a[G_OTW][(o * 64 + i) * 7 + dt];
  }
  for (int i = tid; i < 4096; i += 256) Wdt2L[i] = (_Float16)p.a[G_DT2W][i];
  __syncthreads();

  for (int t = wave; t < SEG; t += 8) {
    v8f zc[4][2], dc[4][2];
#pragma unroll
    for (int mt = 0; mt < 4; ++mt) { zc[mt][0] = zv; zc[mt][1] = zv; dc[mt][0] = zv; dc[mt][1] = zv; }
    for (int dt = 0; dt < 7; ++dt) {
      int tt = t + dt - 3;
      if (tt < 0 || tt >= SEG) continue;
      for (int kk = 0; kk < 2; ++kk) {
        v16h b0 = ld_b(ybuf + tt * 32, 640, kk * 32, 0, lane);
        v16h b1 = ld_b(ybuf + tt * 32, 640, kk * 32, 16, lane);
#pragma unroll
        for (int mt = 0; mt < 4; ++mt) {
          v16h a = ld_a(WotL, 448, mt * 16, dt * 64 + kk * 32, lane);
          zc[mt][0] = wmma16(a, b0, zc[mt][0]);
          zc[mt][1] = wmma16(a, b1, zc[mt][1]);
        }
      }
    }
    for (int kk = 0; kk < 2; ++kk) {
      v16h b0 = ld_b(ybuf + t * 32, 640, kk * 32, 0, lane);
      v16h b1 = ld_b(ybuf + t * 32, 640, kk * 32, 16, lane);
#pragma unroll
      for (int mt = 0; mt < 4; ++mt) {
        v16h a = ld_a(Wdt2L, 64, mt * 16, kk * 32, lane);
        dc[mt][0] = wmma16(a, b0, dc[mt][0]);
        dc[mt][1] = wmma16(a, b1, dc[mt][1]);
      }
    }
    // h = leaky(bn_dt2 + bn_ot) + tem1 ; max over v<25 via shfl_xor
#pragma unroll
    for (int mt = 0; mt < 4; ++mt)
#pragma unroll
      for (int e = 0; e < 8; ++e) {
        int o = mt * 16 + hi8 + e;
        float tbias = t1[o * SEG + t];
        float z0 = zc[mt][0][e] * prm[PK_OT * 128 + o] + prm[PK_OT * 128 + 64 + o];
        float d0 = dc[mt][0][e] * prm[PK_DT2 * 128 + o] + prm[PK_DT2 * 128 + 64 + o];
        float h0v = lrelu(d0 + z0) + tbias;
        float z1 = zc[mt][1][e] * prm[PK_OT * 128 + o] + prm[PK_OT * 128 + 64 + o];
        float d1v = dc[mt][1][e] * prm[PK_DT2 * 128 + o] + prm[PK_DT2 * 128 + 64 + o];
        float h1v = lrelu(d1v + z1) + tbias;
        if (col + 16 >= NJ) h1v = -3.4e38f;  // pad columns
        float m = fmaxf(h0v, h1v);
        m = fmaxf(m, __shfl_xor(m, 1, 32));
        m = fmaxf(m, __shfl_xor(m, 2, 32));
        m = fmaxf(m, __shfl_xor(m, 4, 32));
        m = fmaxf(m, __shfl_xor(m, 8, 32));
        if (col == 0) mj[o * SEG + t] = m;
      }
  }
  __syncthreads();

  // ================= Phase D: tail CNN + FC (tiny, scalar) =================
  float* c1o = (float*)(smem + C1O_OFF);
  float* c2o = (float*)(smem + C2O_OFF);
  float* m2  = (float*)(smem + M2_OFF);
  for (int idx = tid; idx < 64 * SEG; idx += 256) {
    int o = idx / SEG, t = idx % SEG;
    float acc = p.a[C_C1B][o];
    for (int dk = 0; dk < 3; ++dk) {
      int tt = t + dk - 1;
      if (tt < 0 || tt >= SEG) continue;
      const float* wr = p.a[C_C1W] + o * 192 + dk;  // (o*64+i)*3+dk
      for (int i = 0; i < 64; ++i) acc += wr[i * 3] * mj[i * SEG + tt];
    }
    float sc = p.a[C_BN1_G][o] * rsqrtf(p.a[C_BN1_V][o] + 1e-5f);
    acc = (acc - p.a[C_BN1_M][o]) * sc + p.a[C_BN1_B][o];
    c1o[idx] = fmaxf(acc, 0.f);
  }
  __syncthreads();
  for (int idx = tid; idx < 128 * SEG; idx += 256) {
    int o = idx / SEG, t = idx % SEG;
    float acc = p.a[C_C2B][o];
    const float* wr = p.a[C_C2W] + o * 64;
    for (int i = 0; i < 64; ++i) acc += wr[i] * c1o[i * SEG + t];
    float sc = p.a[C_BN2_G][o] * rsqrtf(p.a[C_BN2_V][o] + 1e-5f);
    acc = (acc - p.a[C_BN2_M][o]) * sc + p.a[C_BN2_B][o];
    c2o[idx] = fmaxf(acc, 0.f);
  }
  __syncthreads();
  if (tid < 128) {
    float m = -3.4e38f;
    for (int t = 0; t < SEG; ++t) m = fmaxf(m, c2o[tid * SEG + t]);
    m2[tid] = m;
  }
  __syncthreads();
  if (tid < 60) {
    float acc = p.a[FC_B][tid];
    for (int pc = 0; pc < 128; ++pc) acc += m2[pc] * p.a[FC_W][pc * 60 + tid];
    out[n * 60 + tid] = acc;
  }
}

// =================== host launcher ===================
extern "C" void kernel_launch(void* const* d_in, const int* in_sizes, int n_in,
                              void* d_out, int out_size, void* d_ws,
                              size_t ws_size, hipStream_t stream) {
  (void)in_sizes; (void)out_size; (void)ws_size;
  P p;
  for (int i = 0; i < N_IN; ++i) p.a[i] = (const float*)d_in[i < n_in ? i : 0];

  float* wsf = (float*)d_ws;                              // spa1 [0,400), tem1 [512,1792)
  _Float16* h0 = (_Float16*)((char*)d_ws + 8192);         // f16 [512][32][20][32]
  float* out = (float*)d_out;

  k_precompute<<<1, 64, 0, stream>>>(p, wsf);
  k_embed<<<NSAMP, 512, 0, stream>>>(p, wsf, h0);
  (void)hipFuncSetAttribute((const void*)k_gat,
                            hipFuncAttributeMaxDynamicSharedMemorySize,
                            (int)SMEM_TOTAL);
  k_gat<<<NSAMP, 256, SMEM_TOTAL, stream>>>(p, wsf, h0, out);
}
